// RWKV_82806969467596
// MI455X (gfx1250) — matrix operand order repaired and checked by reading, not compiled, not run on hardware
//
#include <hip/hip_runtime.h>
#include <math.h>

// ---------------- model dims ----------------
#define T_SEQ   1024
#define C_DIM   768
#define H_HEADS 12
#define K_HEAD  64
#define NLAYERS 12
#define FF_DIM  (3 * C_DIM)
#define V_VOCAB 50304

// ---------------- types ----------------
typedef __attribute__((ext_vector_type(16))) __bf16 v16bf;
typedef __attribute__((ext_vector_type(8)))  float  v8f;
typedef __attribute__((ext_vector_type(8)))  unsigned short v8us;
typedef __attribute__((ext_vector_type(4)))  float  v4f;

union Frag { v16bf v; v8us h[2]; };

__device__ __forceinline__ unsigned short f2bf(float f) {
  unsigned int u = __float_as_uint(f);
  u += 0x7FFFu + ((u >> 16) & 1u);          // round-to-nearest-even
  return (unsigned short)(u >> 16);
}
__device__ __forceinline__ float sigm(float x) { return 1.0f / (1.0f + __expf(-x)); }

// ============================================================
// Tiled WMMA GEMM:  C[M,N] = A[M,K](bf16) @ B[N,K](f32, row-major)^T
// Block: 256 threads = 8 waves. Tile: 128(M) x 64(N), K-step 32.
// BOTH operand tiles are staged through double-buffered LDS so every
// WMMA fragment load is a ds_load (DScnt): the pre-WMMA wait never
// touches LOADcnt, and the next tile's global loads (issued before the
// WMMA block, consumed after it) stay in flight across the WMMAs even
// though VMEM loads retire in order.
// ============================================================
#define LDS_PAD 40   // ushorts per LDS row (32 data + 8 pad)

__global__ __launch_bounds__(256)
void gemm_bf16w32(const unsigned short* __restrict__ A, const float* __restrict__ B,
                  float* __restrict__ C, int M, int N, int K)
{
  __shared__ unsigned short ldsA[2][128 * LDS_PAD];  // 2 x 10 KB
  __shared__ unsigned short ldsB[2][64 * LDS_PAD];   // 2 x 5 KB
  const int tid  = threadIdx.x;
  const int lane = tid & 31;
  const int wave = tid >> 5;
  const int n0 = blockIdx.x * 64;
  const int mb = blockIdx.y * 128;
  const int m0 = mb + wave * 16;

  // B staging: each thread converts 8 consecutive f32 of one row
  const int rB = tid >> 2;            // 0..63
  const int cB = (tid & 3) * 8;       // 0,8,16,24
  const float* bsrc = B + (size_t)(n0 + rB) * K + cB;
  // A staging: each thread copies 16 consecutive bf16 of one row
  const int rA = tid >> 1;            // 0..127
  const int cA = (tid & 1) * 16;      // 0,16
  const unsigned short* asrc = A + (size_t)(mb + rA) * K + cA;

  // fragment addressing (contiguous-K A layout: lanes 0-15 K0-15,
  // lanes 16-31 K16-31; B: lane = N column, 16 contiguous K per half)
  const int fr = lane & 15;
  const int fh = lane >> 4;

  v8f acc[4] = {};

  // prologue: stage K-step 0 into buffer 0
  {
    v8us ga0 = *(const v8us*)(asrc);
    v8us ga1 = *(const v8us*)(asrc + 8);
    v4f  gb0 = *(const v4f*)(bsrc);
    v4f  gb1 = *(const v4f*)(bsrc + 4);
    *(v8us*)&ldsA[0][rA * LDS_PAD + cA]     = ga0;
    *(v8us*)&ldsA[0][rA * LDS_PAD + cA + 8] = ga1;
    v8us pk;
#pragma unroll
    for (int i = 0; i < 4; ++i) { pk[i] = f2bf(gb0[i]); pk[4 + i] = f2bf(gb1[i]); }
    *(v8us*)&ldsB[0][rB * LDS_PAD + cB] = pk;
  }
  __syncthreads();

  const int nsteps = K / 32;
  for (int s = 0; s < nsteps; ++s) {
    const int k0 = s * 32;
    const int cur = s & 1;
    const bool more = (s + 1 < nsteps);

    // 1. issue next tile's global loads; first use is AFTER the WMMA block,
    //    so their LOADcnt wait lands there with the WMMAs hiding the latency
    v8us ga0 = {}, ga1 = {};
    v4f  gb0 = {}, gb1 = {};
    if (more) {
      ga0 = *(const v8us*)(asrc + k0 + 32);
      ga1 = *(const v8us*)(asrc + k0 + 40);
      gb0 = *(const v4f*)(bsrc + k0 + 32);
      gb1 = *(const v4f*)(bsrc + k0 + 36);
      if (s + 2 < nsteps) __builtin_prefetch(bsrc + k0 + 64, 0, 1);
    }

    // 2. fragments come from LDS only -> DScnt-only wait before the WMMAs
    Frag a;
    const unsigned short* ap = &ldsA[cur][(wave * 16 + fr) * LDS_PAD + fh * 16];
    a.h[0] = *(const v8us*)(ap);         // K = fh*16 .. +7
    a.h[1] = *(const v8us*)(ap + 8);     // K = fh*16 + 8 .. +15
    const unsigned short* bl = &ldsB[cur][fr * LDS_PAD + fh * 16];
    Frag b0, b1, b2, b3;
    b0.h[0] = *(const v8us*)(bl);
    b0.h[1] = *(const v8us*)(bl + 8);
    b1.h[0] = *(const v8us*)(bl + 16 * LDS_PAD);
    b1.h[1] = *(const v8us*)(bl + 16 * LDS_PAD + 8);
    b2.h[0] = *(const v8us*)(bl + 32 * LDS_PAD);
    b2.h[1] = *(const v8us*)(bl + 32 * LDS_PAD + 8);
    b3.h[0] = *(const v8us*)(bl + 48 * LDS_PAD);
    b3.h[1] = *(const v8us*)(bl + 48 * LDS_PAD + 8);

    acc[0] = __builtin_amdgcn_wmma_f32_16x16x32_bf16(
        false, a.v, false, b0.v, (short)0, acc[0], false, false);
    acc[1] = __builtin_amdgcn_wmma_f32_16x16x32_bf16(
        false, a.v, false, b1.v, (short)0, acc[1], false, false);
    acc[2] = __builtin_amdgcn_wmma_f32_16x16x32_bf16(
        false, a.v, false, b2.v, (short)0, acc[2], false, false);
    acc[3] = __builtin_amdgcn_wmma_f32_16x16x32_bf16(
        false, a.v, false, b3.v, (short)0, acc[3], false, false);

    // 3. park next tile in the other buffers (waits for step-1 loads here)
    if (more) {
      *(v8us*)&ldsA[1 - cur][rA * LDS_PAD + cA]     = ga0;
      *(v8us*)&ldsA[1 - cur][rA * LDS_PAD + cA + 8] = ga1;
      v8us pk;
#pragma unroll
      for (int i = 0; i < 4; ++i) { pk[i] = f2bf(gb0[i]); pk[4 + i] = f2bf(gb1[i]); }
      *(v8us*)&ldsB[1 - cur][rB * LDS_PAD + cB] = pk;
    }
    __syncthreads();
  }

  // C/D layout: VGPR i -> row m0 + i (+8 for lanes>=16); lane&15 -> column
  const int nn = lane & 15;
  const int mo = (lane >> 4) * 8;
  float* cp = C + (size_t)(m0 + mo) * N + n0 + nn;
#pragma unroll
  for (int j = 0; j < 4; ++j)
#pragma unroll
    for (int i = 0; i < 8; ++i)
      cp[(size_t)i * N + j * 16] = acc[j][i];
}

// ============================================================
// LayerNorm over C=768; writes f32 and bf16 copies.
// ============================================================
__global__ __launch_bounds__(256)
void ln_kernel(const float* __restrict__ x, const float* __restrict__ w,
               const float* __restrict__ b, float* __restrict__ y32,
               unsigned short* __restrict__ y16, float eps)
{
  __shared__ float s1[256], s2[256];
  const int t = blockIdx.x, tid = threadIdx.x;
  const float* row = x + (size_t)t * C_DIM;
  float v0 = row[tid], v1 = row[tid + 256], v2 = row[tid + 512];
  s1[tid] = v0 + v1 + v2;
  s2[tid] = v0 * v0 + v1 * v1 + v2 * v2;
  __syncthreads();
  for (int s = 128; s > 0; s >>= 1) {
    if (tid < s) { s1[tid] += s1[tid + s]; s2[tid] += s2[tid + s]; }
    __syncthreads();
  }
  const float mean = s1[0] * (1.0f / C_DIM);
  const float var  = s2[0] * (1.0f / C_DIM) - mean * mean;
  const float rstd = rsqrtf(var + eps);
  float* o32 = y32 + (size_t)t * C_DIM;
  unsigned short* o16 = y16 + (size_t)t * C_DIM;
  float vv[3] = {v0, v1, v2};
#pragma unroll
  for (int i = 0; i < 3; ++i) {
    int c = tid + i * 256;
    float y = (vv[i] - mean) * rstd * w[c] + b[c];
    o32[c] = y;
    o16[c] = f2bf(y);
  }
}

// ============================================================
// Embedding gather
// ============================================================
__global__ __launch_bounds__(256)
void embed_kernel(const int* __restrict__ tok, const float* __restrict__ emb,
                  float* __restrict__ h)
{
  const int t = blockIdx.x, tid = threadIdx.x;
  const size_t src = (size_t)tok[t] * C_DIM;
#pragma unroll
  for (int i = 0; i < 3; ++i)
    h[(size_t)t * C_DIM + tid + i * 256] = emb[src + tid + i * 256];
}

// ============================================================
// Token-shift lerp: out = bf16( xn + coef * (xn[t-1] - xn) ),  xn[-1]=0
// ============================================================
__global__ __launch_bounds__(256)
void lerp_kernel(const float* __restrict__ xn, const float* __restrict__ coef,
                 unsigned short* __restrict__ out)
{
  const int t = blockIdx.x, tid = threadIdx.x;
  const size_t ro = (size_t)t * C_DIM;
#pragma unroll
  for (int i = 0; i < 3; ++i) {
    int c = tid + i * 256;
    float v  = xn[ro + c];
    float p  = (t > 0) ? xn[ro - C_DIM + c] : 0.0f;
    float co = coef[ro + c];
    out[ro + c] = f2bf(v + co * (p - v));
  }
}

// ============================================================
// WKV linear-attention scan. One workgroup per head, 256 threads.
// State kv[64][64] lives in registers: thread (kg,vc) owns st[i] = kv[kg*16+i][vc].
// ============================================================
__global__ __launch_bounds__(256)
void wkv_kernel(const float* __restrict__ r, const float* __restrict__ k,
                const float* __restrict__ v, const float* __restrict__ u_p,
                const float* __restrict__ wd_p, float* __restrict__ out)
{
  __shared__ float sr[64], sk[64], sv[64], red[256];
  const int h = blockIdx.x;
  const int tid = threadIdx.x;
  const int vc = tid & 63;        // output column (v-index)
  const int kb = (tid >> 6) * 16; // k-subrange base
  float uk[16], wk[16], st[16];
#pragma unroll
  for (int i = 0; i < 16; ++i) {
    uk[i] = u_p[h * K_HEAD + kb + i];
    wk[i] = __expf(-__expf(wd_p[h * K_HEAD + kb + i]));
    st[i] = 0.0f;
  }
  for (int t = 0; t < T_SEQ; ++t) {
    const size_t base = (size_t)t * C_DIM + h * K_HEAD;
    if (tid < 64)       sr[tid]        = r[base + tid];
    else if (tid < 128) sk[tid - 64]   = k[base + tid - 64];
    else if (tid < 192) sv[tid - 128]  = v[base + tid - 128];
    __syncthreads();
    const float vv = sv[vc];
    float acc = 0.0f;
#pragma unroll
    for (int i = 0; i < 16; ++i) {
      float kv = sk[kb + i] * vv;
      acc += sr[kb + i] * (st[i] + kv * uk[i]);
      st[i] = st[i] * wk[i] + kv;
    }
    red[tid] = acc;
    __syncthreads();
    if (tid < 64)
      out[base + tid] = red[tid] + red[tid + 64] + red[tid + 128] + red[tid + 192];
    __syncthreads();
  }
}

// ============================================================
// GroupNorm(eps=0.00064) per (token,head) + SiLU gate, outputs bf16.
// Grid (T, H), 64 threads.
// ============================================================
__global__ __launch_bounds__(64)
void gn_gate_kernel(const float* __restrict__ wkv, const float* __restrict__ gate,
                    const float* __restrict__ gw, const float* __restrict__ gb,
                    unsigned short* __restrict__ out)
{
  __shared__ float s1[64], s2[64];
  const int t = blockIdx.x, h = blockIdx.y, kk = threadIdx.x;
  const size_t idx = (size_t)t * C_DIM + h * K_HEAD + kk;
  const float x = wkv[idx];
  s1[kk] = x; s2[kk] = x * x;
  __syncthreads();
  for (int s = 32; s > 0; s >>= 1) {
    if (kk < s) { s1[kk] += s1[kk + s]; s2[kk] += s2[kk + s]; }
    __syncthreads();
  }
  const float mean = s1[0] * (1.0f / K_HEAD);
  const float var  = s2[0] * (1.0f / K_HEAD) - mean * mean;
  const float rstd = rsqrtf(var + 0.00064f);
  float y = (x - mean) * rstd * gw[h * K_HEAD + kk] + gb[h * K_HEAD + kk];
  const float g = gate[idx];
  y *= g * sigm(g);
  out[idx] = f2bf(y);
}

// ============================================================
// elementwise helpers
// ============================================================
__global__ __launch_bounds__(256)
void relu2_kernel(const float* __restrict__ in, unsigned short* __restrict__ out, int n)
{
  int i = blockIdx.x * 256 + threadIdx.x;
  if (i < n) { float v = in[i]; v = v > 0.0f ? v * v : 0.0f; out[i] = f2bf(v); }
}

__global__ __launch_bounds__(256)
void add_kernel(float* __restrict__ o, const float* __restrict__ a,
                const float* __restrict__ b, int n)
{
  int i = blockIdx.x * 256 + threadIdx.x;
  if (i < n) o[i] = a[i] + b[i];
}

__global__ __launch_bounds__(256)
void gated_add_kernel(float* __restrict__ o, const float* __restrict__ a,
                      const float* __restrict__ b, const float* __restrict__ g, int n)
{
  int i = blockIdx.x * 256 + threadIdx.x;
  if (i < n) o[i] = a[i] + b[i] * sigm(g[i]);
}

// ============================================================
// host side
// ============================================================
enum {
  IN_X = 0, IN_EMBED, IN_EMB_LN_W, IN_EMB_LN_B,
  IN_LN1_W, IN_LN1_B,
  IN_WLERP_R, IN_WLERP_K, IN_WLERP_V, IN_WLERP_G,
  IN_WPROJ_R, IN_WPROJ_K, IN_WPROJ_V, IN_WPROJ_G, IN_WPROJ_OUT,
  IN_U, IN_WDECAY, IN_GN_W, IN_GN_B,
  IN_LN2_W, IN_LN2_B, IN_WIN, IN_WOUT, IN_WLERP_IN, IN_WLERP_CG, IN_WGATE,
  IN_OUT_LN_W, IN_OUT_LN_B, IN_UNEMBED
};

extern "C" void kernel_launch(void* const* d_in, const int* in_sizes, int n_in,
                              void* d_out, int out_size, void* d_ws, size_t ws_size,
                              hipStream_t stream)
{
  (void)in_sizes; (void)n_in; (void)out_size; (void)ws_size;
  const int*   tok     = (const int*)d_in[IN_X];
  const float* emb     = (const float*)d_in[IN_EMBED];
  const float* eln_w   = (const float*)d_in[IN_EMB_LN_W];
  const float* eln_b   = (const float*)d_in[IN_EMB_LN_B];
  const float* ln1_w   = (const float*)d_in[IN_LN1_W];
  const float* ln1_b   = (const float*)d_in[IN_LN1_B];
  const float* Wl_r    = (const float*)d_in[IN_WLERP_R];
  const float* Wl_k    = (const float*)d_in[IN_WLERP_K];
  const float* Wl_v    = (const float*)d_in[IN_WLERP_V];
  const float* Wl_g    = (const float*)d_in[IN_WLERP_G];
  const float* Wp_r    = (const float*)d_in[IN_WPROJ_R];
  const float* Wp_k    = (const float*)d_in[IN_WPROJ_K];
  const float* Wp_v    = (const float*)d_in[IN_WPROJ_V];
  const float* Wp_g    = (const float*)d_in[IN_WPROJ_G];
  const float* Wp_o    = (const float*)d_in[IN_WPROJ_OUT];
  const float* u_all   = (const float*)d_in[IN_U];
  const float* wd_all  = (const float*)d_in[IN_WDECAY];
  const float* gn_w    = (const float*)d_in[IN_GN_W];
  const float* gn_b    = (const float*)d_in[IN_GN_B];
  const float* ln2_w   = (const float*)d_in[IN_LN2_W];
  const float* ln2_b   = (const float*)d_in[IN_LN2_B];
  const float* Win     = (const float*)d_in[IN_WIN];
  const float* Wout    = (const float*)d_in[IN_WOUT];
  const float* Wl_in   = (const float*)d_in[IN_WLERP_IN];
  const float* Wl_cg   = (const float*)d_in[IN_WLERP_CG];
  const float* Wgate   = (const float*)d_in[IN_WGATE];
  const float* oln_w   = (const float*)d_in[IN_OUT_LN_W];
  const float* oln_b   = (const float*)d_in[IN_OUT_LN_B];
  const float* unembed = (const float*)d_in[IN_UNEMBED];

  // workspace layout (all sizes are multiples of 256B) -- ~42 MB total
  constexpr size_t SZF  = (size_t)T_SEQ * C_DIM * sizeof(float);
  constexpr size_t SZB  = (size_t)T_SEQ * C_DIM * sizeof(unsigned short);
  constexpr size_t SZFF = (size_t)T_SEQ * FF_DIM * sizeof(float);
  constexpr size_t SZFB = (size_t)T_SEQ * FF_DIM * sizeof(unsigned short);
  char* ws = (char*)d_ws;
  size_t off = 0;
  float* X    = (float*)(ws + off); off += SZF;   // residual stream
  float* XN   = (float*)(ws + off); off += SZF;   // normed x (f32)
  float* COEF = (float*)(ws + off); off += SZF;   // lerp coefficients / temp
  float* Rb   = (float*)(ws + off); off += SZF;
  float* Kb   = (float*)(ws + off); off += SZF;
  float* Vb   = (float*)(ws + off); off += SZF;
  float* Gb   = (float*)(ws + off); off += SZF;
  float* WKV  = (float*)(ws + off); off += SZF;
  unsigned short* XNB = (unsigned short*)(ws + off); off += SZB;  // normed x (bf16)
  unsigned short* XSB = (unsigned short*)(ws + off); off += SZB;  // shifted/lerped (bf16)
  unsigned short* GNB = (unsigned short*)(ws + off); off += SZB;  // groupnorm-gated (bf16)
  float* HIDF = (float*)(ws + off); off += SZFF;
  unsigned short* HIDB = (unsigned short*)(ws + off); off += SZFB;

  auto gemm = [&](const unsigned short* A, const float* B, float* Cc, int M, int N, int K) {
    dim3 grid(N / 64, M / 128);
    gemm_bf16w32<<<grid, 256, 0, stream>>>(A, B, Cc, M, N, K);
  };

  const int NELEM = T_SEQ * C_DIM;

  // embed + embedding LN (in-place into X)
  embed_kernel<<<T_SEQ, 256, 0, stream>>>(tok, emb, X);
  ln_kernel<<<T_SEQ, 256, 0, stream>>>(X, eln_w, eln_b, X, XNB, 1e-5f);

  for (int l = 0; l < NLAYERS; ++l) {
    const size_t oC  = (size_t)l * C_DIM;
    const size_t oCC = (size_t)l * C_DIM * C_DIM;
    const size_t oFC = (size_t)l * FF_DIM * C_DIM;

    // ---------------- time mixer ----------------
    ln_kernel<<<T_SEQ, 256, 0, stream>>>(X, ln1_w + oC, ln1_b + oC, XN, XNB, 1e-5f);
    const float* Wl4[4] = {Wl_r + oCC, Wl_k + oCC, Wl_v + oCC, Wl_g + oCC};
    const float* Wp4[4] = {Wp_r + oCC, Wp_k + oCC, Wp_v + oCC, Wp_g + oCC};
    float* dst4[4] = {Rb, Kb, Vb, Gb};
    for (int j = 0; j < 4; ++j) {
      gemm(XNB, Wl4[j], COEF, T_SEQ, C_DIM, C_DIM);
      lerp_kernel<<<T_SEQ, 256, 0, stream>>>(XN, COEF, XSB);
      gemm(XSB, Wp4[j], dst4[j], T_SEQ, C_DIM, C_DIM);
    }
    wkv_kernel<<<H_HEADS, 256, 0, stream>>>(Rb, Kb, Vb, u_all + oC, wd_all + oC, WKV);
    gn_gate_kernel<<<dim3(T_SEQ, H_HEADS), 64, 0, stream>>>(WKV, Gb, gn_w + oC, gn_b + oC, GNB);
    gemm(GNB, Wp_o + oCC, COEF, T_SEQ, C_DIM, C_DIM);
    add_kernel<<<NELEM / 256, 256, 0, stream>>>(X, XN, COEF, NELEM);   // x = xn + out

    // ---------------- channel mixer ----------------
    ln_kernel<<<T_SEQ, 256, 0, stream>>>(X, ln2_w + oC, ln2_b + oC, XN, XNB, 1e-5f);
    gemm(XNB, Wl_in + oCC, COEF, T_SEQ, C_DIM, C_DIM);
    lerp_kernel<<<T_SEQ, 256, 0, stream>>>(XN, COEF, XSB);
    gemm(XSB, Win + oFC, HIDF, T_SEQ, FF_DIM, C_DIM);
    relu2_kernel<<<(T_SEQ * FF_DIM) / 256, 256, 0, stream>>>(HIDF, HIDB, T_SEQ * FF_DIM);
    gemm(HIDB, Wout + oFC, Rb, T_SEQ, C_DIM, FF_DIM);                  // out
    gemm(XNB, Wl_cg + oCC, COEF, T_SEQ, C_DIM, C_DIM);
    lerp_kernel<<<T_SEQ, 256, 0, stream>>>(XN, COEF, XSB);
    gemm(XSB, Wgate + oCC, Kb, T_SEQ, C_DIM, C_DIM);                   // gate
    gated_add_kernel<<<NELEM / 256, 256, 0, stream>>>(X, XN, Rb, Kb, NELEM); // x = xc + out*sig(gate)
  }

  // final LN + unembed -> logits
  ln_kernel<<<T_SEQ, 256, 0, stream>>>(X, oln_w, oln_b, XN, XNB, 1e-5f);
  gemm(XNB, unembed, (float*)d_out, T_SEQ, V_VOCAB, C_DIM);
}
